// MultiHeadSelfAttention_68771016343811
// MI455X (gfx1250) — matrix operand order repaired and checked by reading, not compiled, not run on hardware
//
#include <hip/hip_runtime.h>
#include <hip/hip_bf16.h>

// ---------------------------------------------------------------------------
// MHA (B=2, S=2048, D=1024, H=16, DH=64) for gfx1250 (MI455X), wave32 + WMMA.
// Pipeline: 3x QKV GEMM (f32->bf16, WMMA bf16/f32-acc) -> RoPE (f32 VALU)
//           -> flash attention (WMMA + online softmax) -> output GEMM (WMMA).
// GEMMs use 128x128 block tiles (8 WMMA / wave / K-step) with packed-b32
// transposed weight staging; all tensors stay L2-resident (~60MB << 192MB).
// ---------------------------------------------------------------------------

typedef __attribute__((ext_vector_type(16))) __bf16 v16bf;
typedef __attribute__((ext_vector_type(8)))  __bf16 v8bf;
typedef __attribute__((ext_vector_type(8)))  float  v8f;

#define C_B   2
#define C_S   2048
#define C_D   1024
#define C_H   16
#define C_DH  64

__device__ __forceinline__ v8f wmma_bf16(v16bf a, v16bf b, v8f c) {
    return __builtin_amdgcn_wmma_f32_16x16x32_bf16(false, a, false, b,
                                                   (short)0, c, false, false);
}

// A fragment (16x32 bf16): lane = m + 16*hi ; K(e) = (e&7) + 8*hi + 16*(e>>3)
__device__ __forceinline__ v16bf frag_a(const __bf16* t, int stride, int row,
                                        int hi, int kb) {
    const __bf16* p = t + row * stride + kb + hi * 8;
    v8bf lo = *(const v8bf*)p;
    v8bf hh = *(const v8bf*)(p + 16);
    return __builtin_shufflevector(lo, hh, 0,1,2,3,4,5,6,7,8,9,10,11,12,13,14,15);
}

// B fragment (32x16 bf16): lane = n + 16*hi ; K(e) = (e&7) + 8*hi + 16*(e>>3)
__device__ __forceinline__ v16bf frag_b(const __bf16* t, int stride, int row,
                                        int hi, int kb) {
    const __bf16* p = t + row * stride + kb + hi * 8;
    v8bf lo = *(const v8bf*)p;
    v8bf hh = *(const v8bf*)(p + 16);
    return __builtin_shufflevector(lo, hh, 0,1,2,3,4,5,6,7,8,9,10,11,12,13,14,15);
}

// ---------------------------------------------------------------------------
// Kernel 1: out[b,h,s,dh] (bf16) = X[m,k] @ W[k,n] + bias[n]; m=(b,s), n=(h,dh)
// 128x128 tile, BK=32, 8 waves: wave = (mt-pair = (w&3)*2, n-group = (w>>2)*4)
// ---------------------------------------------------------------------------
__global__ __launch_bounds__(256) void k_gemm_qkv(
    const float* __restrict__ X, const float* __restrict__ W,
    const float* __restrict__ bias, __bf16* __restrict__ out)
{
    __shared__ __bf16 lx[128 * 40];   // X tile rows, padded stride 40
    __shared__ __bf16 lw[128 * 40];   // W tile transposed: lw[n][k]
    const int tid  = threadIdx.x;
    const int wave = tid >> 5, lane = tid & 31;
    const int hi = lane >> 4, ln = lane & 15;
    const int m0 = blockIdx.x * 128;
    const int n0 = blockIdx.y * 128;
    const int mt0 = (wave & 3) * 2;
    const int ng  = (wave >> 2) * 4;

    v8f acc[2][4] = {};

    for (int k0 = 0; k0 < C_D; k0 += 32) {
        __syncthreads();
        {   // X tile 128x32 f32 -> bf16, b128 LDS stores
            int row = tid >> 1, c0 = (tid & 1) * 16;
            const float* g = X + (size_t)(m0 + row) * C_D + k0 + c0;
            v8bf o0, o1;
            #pragma unroll
            for (int j = 0; j < 8; ++j) o0[j] = (__bf16)g[j];
            #pragma unroll
            for (int j = 0; j < 8; ++j) o1[j] = (__bf16)g[8 + j];
            *(v8bf*)(lx + row * 40 + c0)     = o0;
            *(v8bf*)(lx + row * 40 + c0 + 8) = o1;
        }
        {   // W tile 32x128 -> transposed lw[n][k]; k-pairs packed as b32
            int kp = tid >> 4;             // k-pair 0..15 -> rows 2kp, 2kp+1
            int n0l = (tid & 15) * 8;
            const float* g0 = W + (size_t)(k0 + 2 * kp)     * C_D + n0 + n0l;
            const float* g1 = W + (size_t)(k0 + 2 * kp + 1) * C_D + n0 + n0l;
            unsigned* lwu = (unsigned*)lw;     // row stride 20 uints (= 40 bf16)
            #pragma unroll
            for (int j = 0; j < 8; ++j) {
                union { unsigned u; __bf16 h[2]; } pk;
                pk.h[0] = (__bf16)g0[j];
                pk.h[1] = (__bf16)g1[j];
                lwu[(n0l + j) * 20 + kp] = pk.u;
            }
        }
        __syncthreads();
        v16bf a0 = frag_a(lx, 40, (mt0    ) * 16 + ln, hi, 0);
        v16bf a1 = frag_a(lx, 40, (mt0 + 1) * 16 + ln, hi, 0);
        #pragma unroll
        for (int j = 0; j < 4; ++j) {
            v16bf bj = frag_b(lw, 40, (ng + j) * 16 + ln, hi, 0);
            acc[0][j] = wmma_bf16(a0, bj, acc[0][j]);
            acc[1][j] = wmma_bf16(a1, bj, acc[1][j]);
        }
    }

    #pragma unroll
    for (int j = 0; j < 4; ++j) {
        const int n = n0 + (ng + j) * 16 + ln;
        const float bvl = bias[n];
        const int hh = n >> 6, dd = n & 63;
        #pragma unroll
        for (int i = 0; i < 2; ++i) {
            #pragma unroll
            for (int r = 0; r < 8; ++r) {
                int m = m0 + (mt0 + i) * 16 + hi * 8 + r;
                int bb = m >> 11, ss = m & (C_S - 1);
                out[(((size_t)bb * C_H + hh) * C_S + ss) * C_DH + dd] =
                    (__bf16)(acc[i][j][r] + bvl);
            }
        }
    }
}

// ---------------------------------------------------------------------------
// Kernel 2: RoPE in-place on Q and K, layout [B,H,S,DH], pairs (i, i+32)
// ---------------------------------------------------------------------------
__global__ void k_rope(__bf16* __restrict__ q, __bf16* __restrict__ k)
{
    int t = blockIdx.x * blockDim.x + threadIdx.x;
    if (t >= C_B * C_H * C_S * 32) return;
    int i  = t & 31;
    int s  = (t >> 5) & (C_S - 1);
    int bh = t >> 16;                                   // 2048*32 = 65536
    float inv = __expf(-(float)i * (9.210340371976184f / 32.0f)); // 10000^(-i/32)
    float ang = (float)s * inv;
    float c = cosf(ang), sn = sinf(ang);
    size_t base = ((size_t)bh * C_S + s) * C_DH + i;
    float q1 = (float)q[base], q2 = (float)q[base + 32];
    q[base]      = (__bf16)(q1 * c - q2 * sn);
    q[base + 32] = (__bf16)(q1 * sn + q2 * c);
    float k1 = (float)k[base], k2 = (float)k[base + 32];
    k[base]      = (__bf16)(k1 * c - k2 * sn);
    k[base + 32] = (__bf16)(k1 * sn + k2 * c);
}

// ---------------------------------------------------------------------------
// Kernel 3: causal flash attention. Block = (128 q-rows) x (b,h); 8 waves,
// each wave owns 16 q-rows. K/V chunk of 32 keys staged in LDS per block.
// ---------------------------------------------------------------------------
__global__ __launch_bounds__(256) void k_attn(
    const __bf16* __restrict__ Q, const __bf16* __restrict__ K,
    const __bf16* __restrict__ V, __bf16* __restrict__ ctx)
{
    __shared__ __bf16 lq[128 * 72];      // Q rows for block (stride 72)
    __shared__ __bf16 lk[32 * 72];       // K chunk [kc][d]
    __shared__ __bf16 lv[64 * 40];       // V chunk transposed [d][kc]
    __shared__ __bf16 lp[8 * 16 * 40];   // per-wave P tile [16][32]

    const int tid  = threadIdx.x;
    const int wave = tid >> 5, lane = tid & 31;
    const int hi = lane >> 4, ln = lane & 15;
    const int bh = blockIdx.y;
    const int b  = bh >> 4, h = bh & 15;
    const int q0 = blockIdx.x * 128;
    const int qw = q0 + wave * 16;

    // stage 128x64 Q rows (coalesced 16B loads)
    const __bf16* Qg = Q + ((size_t)bh * C_S + q0) * C_DH;
    #pragma unroll
    for (int it = 0; it < 2; ++it) {
        int g = tid + it * 256;               // 512 groups of 16 elems
        int row = g >> 2, c0 = (g & 3) * 16;
        *(v8bf*)(lq + row * 72 + c0)     = *(const v8bf*)(Qg + row * C_DH + c0);
        *(v8bf*)(lq + row * 72 + c0 + 8) = *(const v8bf*)(Qg + row * C_DH + c0 + 8);
    }
    __syncthreads();

    v16bf aq0 = frag_a(lq, 72, wave * 16 + ln, hi, 0);    // d 0..31
    v16bf aq1 = frag_a(lq, 72, wave * 16 + ln, hi, 32);   // d 32..63

    float rmax[8], rsum[8];
    v8f ca0 = {}, ca1 = {}, ca2 = {}, ca3 = {};
    #pragma unroll
    for (int r = 0; r < 8; ++r) { rmax[r] = -__builtin_inff(); rsum[r] = 0.f; }

    const __bf16* Kg = K + (size_t)bh * C_S * C_DH;
    const __bf16* Vg = V + (size_t)bh * C_S * C_DH;
    const int nch = (q0 >> 5) + 4;            // chunks of 32 keys

    for (int ch = 0; ch < nch; ++ch) {
        const int kb = ch * 32;
        __syncthreads();
        {   // K chunk 32x64 -> lk[kc][d]
            int row = tid >> 3, c0 = (tid & 7) * 8;
            *(v8bf*)(lk + row * 72 + c0) =
                *(const v8bf*)(Kg + (size_t)(kb + row) * C_DH + c0);
        }
        if (tid < 128) {   // V chunk 32x64 -> lv[d][kc], kc-pairs packed b32
            int kp = tid >> 3;                 // kc-pair 0..15
            int d0 = (tid & 7) * 8;
            v8bf v0 = *(const v8bf*)(Vg + (size_t)(kb + 2 * kp)     * C_DH + d0);
            v8bf v1 = *(const v8bf*)(Vg + (size_t)(kb + 2 * kp + 1) * C_DH + d0);
            unsigned* lvu = (unsigned*)lv;     // row stride 20 uints
            #pragma unroll
            for (int j = 0; j < 8; ++j) {
                union { unsigned u; __bf16 h[2]; } pk;
                pk.h[0] = v0[j];
                pk.h[1] = v1[j];
                lvu[(d0 + j) * 20 + kp] = pk.u;
            }
        }
        __syncthreads();
        if (kb > qw + 15) continue;           // wave-uniform causal skip

        // ---- scores: 16x32 tile = two 16x16 WMMA outputs, K-dim = DH=64
        v8f s0 = {}, s1 = {};
        {
            v16bf bk0 = frag_b(lk, 72, ln,      hi, 0);
            v16bf bk1 = frag_b(lk, 72, ln,      hi, 32);
            v16bf bk2 = frag_b(lk, 72, 16 + ln, hi, 0);
            v16bf bk3 = frag_b(lk, 72, 16 + ln, hi, 32);
            s0 = wmma_bf16(aq0, bk0, s0);  s0 = wmma_bf16(aq1, bk1, s0);
            s1 = wmma_bf16(aq0, bk2, s1);  s1 = wmma_bf16(aq1, bk3, s1);
        }

        // ---- scale, causal mask, online softmax
        float p0[8], p1[8];
        #pragma unroll
        for (int r = 0; r < 8; ++r) {
            const int qrow = qw + hi * 8 + r;
            float x0 = s0[r] * 0.125f;        // 1/sqrt(64)
            float x1 = s1[r] * 0.125f;
            if (kb + ln > qrow)      x0 = -__builtin_inff();
            if (kb + 16 + ln > qrow) x1 = -__builtin_inff();
            float mx = fmaxf(x0, x1);
            #pragma unroll
            for (int m = 1; m < 16; m <<= 1) mx = fmaxf(mx, __shfl_xor(mx, m, 32));
            float nm = fmaxf(rmax[r], mx);
            float al = __expf(rmax[r] - nm);
            float e0 = __expf(x0 - nm);
            float e1 = __expf(x1 - nm);
            float sm = e0 + e1;
            #pragma unroll
            for (int m = 1; m < 16; m <<= 1) sm += __shfl_xor(sm, m, 32);
            rsum[r] = rsum[r] * al + sm;
            rmax[r] = nm;
            ca0[r] *= al; ca1[r] *= al; ca2[r] *= al; ca3[r] *= al;
            p0[r] = e0; p1[r] = e1;
        }

        // ---- P (acc layout) -> wave-private LDS -> A-fragment layout
        #pragma unroll
        for (int r = 0; r < 8; ++r) {
            __bf16* pr = lp + (wave * 16 + hi * 8 + r) * 40;
            pr[ln]      = (__bf16)p0[r];
            pr[16 + ln] = (__bf16)p1[r];
        }
        v16bf ap  = frag_a(lp, 40, wave * 16 + ln, hi, 0);
        v16bf bv0 = frag_b(lv, 40, ln,      hi, 0);
        v16bf bv1 = frag_b(lv, 40, 16 + ln, hi, 0);
        v16bf bv2 = frag_b(lv, 40, 32 + ln, hi, 0);
        v16bf bv3 = frag_b(lv, 40, 48 + ln, hi, 0);
        ca0 = wmma_bf16(ap, bv0, ca0);
        ca1 = wmma_bf16(ap, bv1, ca1);
        ca2 = wmma_bf16(ap, bv2, ca2);
        ca3 = wmma_bf16(ap, bv3, ca3);
    }

    // ---- normalize and store ctx as bf16 [B,S,D]
    #pragma unroll
    for (int r = 0; r < 8; ++r) {
        const int qrow = qw + hi * 8 + r;
        const float inv = 1.0f / rsum[r];
        size_t base = ((size_t)b * C_S + qrow) * C_D + h * C_DH;
        ctx[base + 0  + ln] = (__bf16)(ca0[r] * inv);
        ctx[base + 16 + ln] = (__bf16)(ca1[r] * inv);
        ctx[base + 32 + ln] = (__bf16)(ca2[r] * inv);
        ctx[base + 48 + ln] = (__bf16)(ca3[r] * inv);
    }
}

// ---------------------------------------------------------------------------
// Kernel 4: out[m,n] (f32) = ctx_bf16[m,k] @ Wo[k,n] + bo[n]; 128x128 tile
// ---------------------------------------------------------------------------
__global__ __launch_bounds__(256) void k_gemm_out(
    const __bf16* __restrict__ A, const float* __restrict__ W,
    const float* __restrict__ bias, float* __restrict__ out)
{
    __shared__ __bf16 la[128 * 40];
    __shared__ __bf16 lw[128 * 40];
    const int tid  = threadIdx.x;
    const int wave = tid >> 5, lane = tid & 31;
    const int hi = lane >> 4, ln = lane & 15;
    const int m0 = blockIdx.x * 128;
    const int n0 = blockIdx.y * 128;
    const int mt0 = (wave & 3) * 2;
    const int ng  = (wave >> 2) * 4;

    v8f acc[2][4] = {};

    for (int k0 = 0; k0 < C_D; k0 += 32) {
        __syncthreads();
        {   // A tile 128x32 bf16, two b128 loads/stores per thread
            int row = tid >> 1, c0 = (tid & 1) * 16;
            const __bf16* g = A + (size_t)(m0 + row) * C_D + k0 + c0;
            *(v8bf*)(la + row * 40 + c0)     = *(const v8bf*)g;
            *(v8bf*)(la + row * 40 + c0 + 8) = *(const v8bf*)(g + 8);
        }
        {   // W tile 32x128 -> transposed, packed b32
            int kp = tid >> 4;
            int n0l = (tid & 15) * 8;
            const float* g0 = W + (size_t)(k0 + 2 * kp)     * C_D + n0 + n0l;
            const float* g1 = W + (size_t)(k0 + 2 * kp + 1) * C_D + n0 + n0l;
            unsigned* lwu = (unsigned*)lw;
            #pragma unroll
            for (int j = 0; j < 8; ++j) {
                union { unsigned u; __bf16 h[2]; } pk;
                pk.h[0] = (__bf16)g0[j];
                pk.h[1] = (__bf16)g1[j];
                lwu[(n0l + j) * 20 + kp] = pk.u;
            }
        }
        __syncthreads();
        v16bf a0 = frag_a(la, 40, (mt0    ) * 16 + ln, hi, 0);
        v16bf a1 = frag_a(la, 40, (mt0 + 1) * 16 + ln, hi, 0);
        #pragma unroll
        for (int j = 0; j < 4; ++j) {
            v16bf bj = frag_b(lw, 40, (ng + j) * 16 + ln, hi, 0);
            acc[0][j] = wmma_bf16(a0, bj, acc[0][j]);
            acc[1][j] = wmma_bf16(a1, bj, acc[1][j]);
        }
    }

    #pragma unroll
    for (int j = 0; j < 4; ++j) {
        const int n = n0 + (ng + j) * 16 + ln;
        const float bvl = bias[n];
        #pragma unroll
        for (int i = 0; i < 2; ++i) {
            #pragma unroll
            for (int r = 0; r < 8; ++r) {
                size_t m = (size_t)(m0 + (mt0 + i) * 16 + hi * 8 + r);
                out[m * C_D + n] = acc[i][j][r] + bvl;
            }
        }
    }
}

// ---------------------------------------------------------------------------
extern "C" void kernel_launch(void* const* d_in, const int* in_sizes, int n_in,
                              void* d_out, int out_size, void* d_ws, size_t ws_size,
                              hipStream_t stream) {
    (void)in_sizes; (void)n_in; (void)out_size; (void)ws_size;
    const float* x  = (const float*)d_in[0];
    const float* Wq = (const float*)d_in[1];
    const float* bq = (const float*)d_in[2];
    const float* Wk = (const float*)d_in[3];
    const float* bk = (const float*)d_in[4];
    const float* Wv = (const float*)d_in[5];
    const float* bv = (const float*)d_in[6];
    const float* Wo = (const float*)d_in[7];
    const float* bo = (const float*)d_in[8];

    const size_t NE = (size_t)C_B * C_S * C_D;   // 4M elements
    __bf16* qr = (__bf16*)d_ws;
    __bf16* kr = qr + NE;
    __bf16* vr = kr + NE;
    __bf16* cx = vr + NE;

    dim3 gg(C_B * C_S / 128, C_D / 128);         // (32, 8)
    k_gemm_qkv<<<gg, 256, 0, stream>>>(x, Wq, bq, qr);
    k_gemm_qkv<<<gg, 256, 0, stream>>>(x, Wk, bk, kr);
    k_gemm_qkv<<<gg, 256, 0, stream>>>(x, Wv, bv, vr);

    k_rope<<<(C_B * C_H * C_S * 32) / 256, 256, 0, stream>>>(qr, kr);

    k_attn<<<dim3(C_S / 128, C_B * C_H), 256, 0, stream>>>(qr, kr, vr, cx);

    k_gemm_out<<<gg, 256, 0, stream>>>(cx, Wo, bo, (float*)d_out);
}
